// BasicBlock_36661840839371
// MI455X (gfx1250) — 8-bit matrix operand order repaired and checked by reading, compiled, not run on hardware
//
#include <hip/hip_runtime.h>


typedef __attribute__((ext_vector_type(8))) int v8i;
typedef int v4si __attribute__((vector_size(16)));

#define C_CH 256
#define HW   3136
#define IMGW 56
#define KTOT 2304
#define KBLK 36
#define BM   128
#define BN   128
#define LDST 80

#if __has_builtin(__builtin_amdgcn_global_load_async_to_lds_b128) && \
    __has_builtin(__builtin_amdgcn_s_wait_asynccnt)
#define USE_ASYNC 1
#define GLOBAL_AS(p) ((__attribute__((address_space(1))) v4si*)((void*)(p)))
#define LDS_AS(p) ((__attribute__((address_space(3))) v4si*)((void*)(p)))
#else
#define USE_ASYNC 0
#endif


__global__ void absmax_tanh_kernel(const float* __restrict__ w, int n,
                                   unsigned int* __restrict__ slot) {
  __shared__ float red[256];
  float m = 0.0f;
  for (int i = blockIdx.x * blockDim.x + threadIdx.x; i < n;
       i += gridDim.x * blockDim.x)
    m = fmaxf(m, fabsf(tanhf(w[i])));
  red[threadIdx.x] = m;
  __syncthreads();
  for (int s = 128; s > 0; s >>= 1) {
    if (threadIdx.x < s)
      red[threadIdx.x] = fmaxf(red[threadIdx.x], red[threadIdx.x + s]);
    __syncthreads();
  }
  if (threadIdx.x == 0) atomicMax(slot, __float_as_uint(red[0]));
}

__global__ void weight_pack_kernel(const float* __restrict__ w,
                                   const unsigned int* __restrict__ slot,
                                   signed char* __restrict__ wq) {
  int idx = blockIdx.x * blockDim.x + threadIdx.x;
  int o = idx / KTOT;
  int k = idx - o * KTOT;
  int r = k >> 8;
  int c = k & 255;
  float mx = __uint_as_float(*slot);
  float wt = tanhf(w[(o * C_CH + c) * 9 + r]);
  float wn = wt / (2.0f * mx) + 0.5f;
  int q = (int)rintf(wn * 3.0f);
  q = q < 0 ? 0 : (q > 3 ? 3 : q);
  wq[idx] = (signed char)(2 * q - 3);
}


__global__ void bn_act_quant_kernel(const float* __restrict__ x,
                                    const float* __restrict__ g,
                                    const float* __restrict__ b,
                                    const float* __restrict__ mean,
                                    const float* __restrict__ var,
                                    unsigned char* __restrict__ act) {
  long idx = (long)blockIdx.x * blockDim.x + threadIdx.x;
  int c = (int)(idx & 255);
  long m = idx >> 8;
  int nimg = (int)(m / HW);
  int pix = (int)(m - (long)nimg * HW);
  float inv = g[c] * rsqrtf(var[c] + 1e-5f);
  float add = b[c] - mean[c] * inv;
  float v = x[((long)(nimg * C_CH + c)) * HW + pix] * inv + add;
  v = fminf(fmaxf(v, 0.0f), 1.0f);
  act[idx] = (unsigned char)(int)rintf(v * 3.0f);
}


__device__ __forceinline__ void stage_tile(
    int kb, unsigned char* __restrict__ Abuf, unsigned char* __restrict__ Bbuf,
    const unsigned char* __restrict__ A, const signed char* __restrict__ W,
    int row, int seg, int py, int px, long abase_img, long wrow) {
  const int r = kb >> 2;
  const int c0 = (kb & 3) << 6;
  const int rdiv = r / 3;
  const int yy = py + rdiv - 1;
  const int xx = px + (r - rdiv * 3) - 1;
  const bool inb = (yy >= 0) && (yy < IMGW) && (xx >= 0) && (xx < IMGW);
  unsigned char* adst = Abuf + row * LDST + seg * 32;
  unsigned char* bdst = Bbuf + row * LDST + seg * 32;
  const signed char* wsrc = W + wrow + (long)kb * 64;
  const unsigned char* asrc =
      A + ((abase_img + (long)yy * IMGW + xx) * C_CH + c0 + seg * 32);
#if USE_ASYNC
  __builtin_amdgcn_global_load_async_to_lds_b128(GLOBAL_AS(wsrc),
                                                 LDS_AS(bdst), 0, 0);
  __builtin_amdgcn_global_load_async_to_lds_b128(GLOBAL_AS(wsrc + 16),
                                                 LDS_AS(bdst + 16), 0, 0);
  if (inb) {
    __builtin_amdgcn_global_load_async_to_lds_b128(GLOBAL_AS(asrc),
                                                   LDS_AS(adst), 0, 0);
    __builtin_amdgcn_global_load_async_to_lds_b128(GLOBAL_AS(asrc + 16),
                                                   LDS_AS(adst + 16), 0, 0);
  } else {
    const int4 z = make_int4(0, 0, 0, 0);
    ((int4*)adst)[0] = z;
    ((int4*)adst)[1] = z;
  }
#else
  const int4* ws4 = (const int4*)wsrc;
  int4 b0 = ws4[0], b1 = ws4[1];
  int4 a0, a1;
  if (inb) {
    const int4* as4 = (const int4*)asrc;
    a0 = as4[0];
    a1 = as4[1];
  } else {
    a0 = make_int4(0, 0, 0, 0);
    a1 = a0;
  }
  ((int4*)bdst)[0] = b0;
  ((int4*)bdst)[1] = b1;
  ((int4*)adst)[0] = a0;
  ((int4*)adst)[1] = a1;
#endif
}


template <int EPI>
__global__ void __launch_bounds__(256)
conv3x3_wmma_iu8_kernel(const unsigned char* __restrict__ A,
                        const signed char* __restrict__ W,
                        const float* __restrict__ g2, const float* __restrict__ b2,
                        const float* __restrict__ m2, const float* __restrict__ v2,
                        unsigned char* __restrict__ actout,
                        const float* __restrict__ xres,
                        float* __restrict__ out) {
  __shared__ unsigned char Alds[2][BM * LDST];
  __shared__ unsigned char Blds[2][BN * LDST];
  __shared__ float invs[C_CH], adds[C_CH];

  const int tid = threadIdx.x;
  const int lane = tid & 31;
  const int wid = tid >> 5;
  const int wave_m = wid >> 2;
  const int wave_n = wid & 3;
  const int bm = blockIdx.x;
  const int bn = blockIdx.y;

  if (EPI == 1) {
    float inv = g2[tid] * rsqrtf(v2[tid] + 1e-5f);
    invs[tid] = inv;
    adds[tid] = b2[tid] - m2[tid] * inv;
  }

  const int row = tid >> 1;
  const int seg = tid & 1;
  const int gm = bm * BM + row;
  const int nimg = gm / HW;
  const int pix = gm - nimg * HW;
  const int py = pix / IMGW;
  const int px = pix - py * IMGW;
  const long abase_img = (long)nimg * HW;
  const long wrow = (long)(bn * BN + row) * KTOT + seg * 32;

  v8i acc[4][2] = {};

  const int l16 = lane & 15;
  const int half = lane >> 4;
  const int koff = half * 8;

  stage_tile(0, Alds[0], Blds[0], A, W, row, seg, py, px, abase_img, wrow);

  for (int kb = 0; kb < KBLK; ++kb) {
    const int buf = kb & 1;
#if USE_ASYNC
    __builtin_amdgcn_s_wait_asynccnt(0);
#endif
    __syncthreads();

    if (kb + 1 < KBLK)
      stage_tile(kb + 1, Alds[buf ^ 1], Blds[buf ^ 1], A, W, row, seg, py, px,
                 abase_img, wrow);

    v8i afrag[4];
#pragma unroll
    for (int mt = 0; mt < 4; ++mt) {
      const unsigned char* p =
          &Alds[buf][(wave_m * 64 + mt * 16 + l16) * LDST];
      union { unsigned long long q[4]; v8i v; } ua;
      ua.q[0] = *(const unsigned long long*)(p + 0 + 2 * koff);
      ua.q[1] = *(const unsigned long long*)(p + 8 + 2 * koff);
      ua.q[2] = *(const unsigned long long*)(p + 32 + 2 * koff);
      ua.q[3] = *(const unsigned long long*)(p + 40 + 2 * koff);
      afrag[mt] = ua.v;
    }
    v8i bfrag[2];
#pragma unroll
    for (int nt = 0; nt < 2; ++nt) {
      const unsigned char* p =
          &Blds[buf][(wave_n * 32 + nt * 16 + l16) * LDST];
      union { int4 q[2]; v8i v; } ub;
      ub.q[0] = *(const int4*)(p + half * 16);
      ub.q[1] = *(const int4*)(p + 32 + half * 16);
      bfrag[nt] = ub.v;
    }
#pragma unroll
    for (int mt = 0; mt < 4; ++mt)
#pragma unroll
      for (int nt = 0; nt < 2; ++nt)
        acc[mt][nt] = __builtin_amdgcn_wmma_i32_16x16x64_iu8(
            false, afrag[mt], true, bfrag[nt], acc[mt][nt], false, false);
  }

  const int ncol = lane & 15;
  const int mo = (lane >> 4) * 8;
  const float s = 1.0f / 9.0f;
#pragma unroll
  for (int mt = 0; mt < 4; ++mt) {
#pragma unroll
    for (int nt = 0; nt < 2; ++nt) {
      const int gn = bn * BN + wave_n * 32 + nt * 16 + ncol;
#pragma unroll
      for (int r = 0; r < 8; ++r) {
        const int gmi = bm * BM + wave_m * 64 + mt * 16 + mo + r;
        const float y = (float)acc[mt][nt][r] * s;
        if (EPI == 1) {
          float t = y * invs[gn] + adds[gn];
          t = fminf(fmaxf(t, 0.0f), 1.0f);
          actout[(long)gmi * C_CH + gn] = (unsigned char)(int)rintf(t * 3.0f);
        } else {
          const int ni = gmi / HW;
          const int pp = gmi - ni * HW;
          const long off = ((long)(ni * C_CH + gn)) * HW + pp;
          out[off] = y + xres[off];
        }
      }
    }
  }
}


extern "C" void kernel_launch(void* const* d_in, const int* in_sizes, int n_in,
                              void* d_out, int out_size, void* d_ws,
                              size_t ws_size, hipStream_t stream) {
  (void)in_sizes; (void)n_in; (void)out_size; (void)ws_size;
  const float* x  = (const float*)d_in[0];
  const float* w1 = (const float*)d_in[1];
  const float* w2 = (const float*)d_in[2];
  const float* g1 = (const float*)d_in[3];
  const float* b1 = (const float*)d_in[4];
  const float* m1 = (const float*)d_in[5];
  const float* v1 = (const float*)d_in[6];
  const float* g2 = (const float*)d_in[7];
  const float* b2 = (const float*)d_in[8];
  const float* m2 = (const float*)d_in[9];
  const float* v2 = (const float*)d_in[10];
  float* out = (float*)d_out;

  char* ws = (char*)d_ws;
  const size_t actBytes = (size_t)100352 * 256;
  unsigned char* act1 = (unsigned char*)ws;
  unsigned char* act2 = act1 + actBytes;
  signed char* wq1 = (signed char*)(act2 + actBytes);
  signed char* wq2 = wq1 + (size_t)C_CH * KTOT;
  unsigned int* slots = (unsigned int*)(wq2 + (size_t)C_CH * KTOT);

  (void)hipMemsetAsync(slots, 0, 2 * sizeof(unsigned int), stream);
  absmax_tanh_kernel<<<256, 256, 0, stream>>>(w1, C_CH * C_CH * 9, slots + 0);
  absmax_tanh_kernel<<<256, 256, 0, stream>>>(w2, C_CH * C_CH * 9, slots + 1);
  weight_pack_kernel<<<KTOT, 256, 0, stream>>>(w1, slots + 0, wq1);
  weight_pack_kernel<<<KTOT, 256, 0, stream>>>(w2, slots + 1, wq2);
  bn_act_quant_kernel<<<100352, 256, 0, stream>>>(x, g1, b1, m1, v1, act1);

  dim3 grid(784, 2);
  conv3x3_wmma_iu8_kernel<1><<<grid, 256, 0, stream>>>(
      act1, wq1, g2, b2, m2, v2, act2, nullptr, nullptr);
  conv3x3_wmma_iu8_kernel<2><<<grid, 256, 0, stream>>>(
      act2, wq2, nullptr, nullptr, nullptr, nullptr, nullptr, x, out);
}
